// MultiHeadSelfAttention_14894946583170
// MI455X (gfx1250) — matrix operand order repaired and checked by reading, not compiled, not run on hardware
//
#include <hip/hip_runtime.h>
#include <hip/hip_bf16.h>
#include <math.h>

// Problem constants (B=2, S=2048, D=1024, H=16, dk=64)
#define BATCH   2
#define SEQ     2048
#define DMODEL  1024
#define HEADS   16
#define DK      64
#define SCALE   0.125f   // 1/sqrt(64)

typedef float  v8f    __attribute__((ext_vector_type(8)));
typedef __bf16 bf16x16 __attribute__((ext_vector_type(16)));

__device__ __forceinline__ __bf16 to_bf16(float f) {
  unsigned int u = __float_as_uint(f);
  unsigned int r = (u + 0x7FFFu + ((u >> 16) & 1u)) >> 16;  // round-to-nearest-even
  unsigned short s = (unsigned short)r;
  return __builtin_bit_cast(__bf16, s);
}
__device__ __forceinline__ float bf16_to_f32(__bf16 h) {
  unsigned short s = __builtin_bit_cast(unsigned short, h);
  return __uint_as_float(((unsigned int)s) << 16);
}

// ---------------------------------------------------------------------------
// WMMA helpers. CDNA5 16x16x32 bf16, f32 accumulate.
// A-frag (16x32, row-major source, leading dim ld):
//   lane l: row m = l&15; half = l>>4.
//   elements 0..15 -> K = half*16 + j
// B-frag (32x16) built from a row-major [N x K] source (i.e. we multiply by
// the source's transpose): lane l: col n = l&15; K = (l>>4)*16 + j.
// C/D (16x16 f32): lane l: col n = l&15; vgpr r -> row m = r + 8*(l>>4).
// ---------------------------------------------------------------------------
__device__ __forceinline__ v8f wmma_bf16(bf16x16 a, bf16x16 b, v8f c) {
  return __builtin_amdgcn_wmma_f32_16x16x32_bf16(
      /*neg_a=*/false, a, /*neg_b=*/false, b,
      /*c_mod=*/(short)0, c, /*reuse_a=*/false, /*reuse_b=*/false);
}

__device__ __forceinline__ bf16x16 load_a_frag(const __bf16* __restrict__ p, int ld) {
  const int lane = threadIdx.x & 31;
  const int m    = lane & 15;
  const int half = lane >> 4;
  const __bf16* r = p + m * ld + half * 16;
  bf16x16 a;
#pragma unroll
  for (int j = 0; j < 8; ++j) a[j]     = r[j];        // 16B load
#pragma unroll
  for (int j = 0; j < 8; ++j) a[8 + j] = r[8 + j];    // 16B load
  return a;
}

__device__ __forceinline__ bf16x16 load_bT_frag(const __bf16* __restrict__ p, int ld) {
  const int lane = threadIdx.x & 31;
  const int n    = lane & 15;
  const __bf16* r = p + n * ld + (lane >> 4) * 16;
  bf16x16 b;
#pragma unroll
  for (int j = 0; j < 16; ++j) b[j] = r[j];           // 32B contiguous
  return b;
}

__device__ __forceinline__ void store_out(float v, float* p)  { *p = v; }
__device__ __forceinline__ void store_out(float v, __bf16* p) { *p = to_bf16(v); }

// ---------------------------------------------------------------------------
// fp32 -> bf16 conversion (grid-stride)
// ---------------------------------------------------------------------------
__global__ void cvt_f32_bf16(const float* __restrict__ s, __bf16* __restrict__ d, int n) {
  for (int i = blockIdx.x * blockDim.x + threadIdx.x; i < n;
       i += gridDim.x * blockDim.x)
    d[i] = to_bf16(s[i]);
}

// ---------------------------------------------------------------------------
// C[m,n] = sum_k A[m,k] * W[n,k]   (y = A · Wᵀ), bf16 inputs, f32 accumulate.
// One wave computes a 32x64 tile: 2 A-frags + 4 B-frags -> 8 WMMAs per K-step
// (2x the WMMA work per fragment load vs a 16x64 tile).
// ---------------------------------------------------------------------------
template <typename OutT>
__global__ void __launch_bounds__(128) gemm_xwt(
    const __bf16* __restrict__ A, const __bf16* __restrict__ W,
    OutT* __restrict__ C, int M, int N, int K) {
  const int lane   = threadIdx.x & 31;
  const int wid    = (blockIdx.x * blockDim.x + threadIdx.x) >> 5;
  const int mtiles = M >> 5;                 // 32-row supertiles
  const int mtile  = wid % mtiles;
  const int nsuper = wid / mtiles;
  const int m0 = mtile * 32;
  const int n0 = nsuper * 64;
  if (n0 >= N) return;

  v8f c00 = {}, c01 = {}, c02 = {}, c03 = {};
  v8f c10 = {}, c11 = {}, c12 = {}, c13 = {};
  for (int k0 = 0; k0 < K; k0 += 32) {
    bf16x16 a0 = load_a_frag(A + (size_t)m0 * K + k0, K);
    bf16x16 a1 = load_a_frag(A + (size_t)(m0 + 16) * K + k0, K);
    bf16x16 b0 = load_bT_frag(W + (size_t)(n0 +  0) * K + k0, K);
    bf16x16 b1 = load_bT_frag(W + (size_t)(n0 + 16) * K + k0, K);
    bf16x16 b2 = load_bT_frag(W + (size_t)(n0 + 32) * K + k0, K);
    bf16x16 b3 = load_bT_frag(W + (size_t)(n0 + 48) * K + k0, K);
    c00 = wmma_bf16(a0, b0, c00);
    c01 = wmma_bf16(a0, b1, c01);
    c02 = wmma_bf16(a0, b2, c02);
    c03 = wmma_bf16(a0, b3, c03);
    c10 = wmma_bf16(a1, b0, c10);
    c11 = wmma_bf16(a1, b1, c11);
    c12 = wmma_bf16(a1, b2, c12);
    c13 = wmma_bf16(a1, b3, c13);
  }

  const int n = lane & 15, half = lane >> 4;
#pragma unroll
  for (int r = 0; r < 8; ++r) {
    const int row0 = m0 + r + 8 * half;
    OutT* cp0 = C + (size_t)row0 * N + n0;
    store_out(c00[r], cp0 +      n);
    store_out(c01[r], cp0 + 16 + n);
    store_out(c02[r], cp0 + 32 + n);
    store_out(c03[r], cp0 + 48 + n);
    OutT* cp1 = cp0 + (size_t)16 * N;
    store_out(c10[r], cp1 +      n);
    store_out(c11[r], cp1 + 16 + n);
    store_out(c12[r], cp1 + 32 + n);
    store_out(c13[r], cp1 + 48 + n);
  }
}

// ---------------------------------------------------------------------------
// RoPE + permute: y[b,s,h*dk+d] (bf16) -> out[b,h,s,d] (bf16), rotated.
// One thread per (b,s,h,i), i in [0,32) handling dims (2i, 2i+1).
// ---------------------------------------------------------------------------
__global__ void rope_permute(const __bf16* __restrict__ y,
                             const float* __restrict__ cosT,
                             const float* __restrict__ sinT,
                             __bf16* __restrict__ out) {
  const int idx = blockIdx.x * blockDim.x + threadIdx.x;  // B*S*H*32 threads
  const int i = idx & 31;
  const int h = (idx >> 5) & 15;
  const int s = (idx >> 9) & (SEQ - 1);
  const int b = idx >> 20;
  const __bf16* yp = y + ((size_t)(b * SEQ + s)) * DMODEL + h * DK + 2 * i;
  const float te = bf16_to_f32(yp[0]);
  const float to = bf16_to_f32(yp[1]);
  const float c  = cosT[s * 32 + i];
  const float sn = sinT[s * 32 + i];
  __bf16* op = out + (((size_t)(b * HEADS + h) * SEQ) + s) * DK + 2 * i;
  op[0] = to_bf16(c * te - sn * to);
  op[1] = to_bf16(sn * te + c * to);
}

// V permute to transposed layout: y[b,s,h*dk+d] -> vT[b,h,d,s]
__global__ void permute_vT(const __bf16* __restrict__ y, __bf16* __restrict__ vT) {
  const int idx = blockIdx.x * blockDim.x + threadIdx.x;  // B*S*D threads
  const int e = idx & (DMODEL - 1);
  const int s = (idx >> 10) & (SEQ - 1);
  const int b = idx >> 21;
  const int d = e & 63, h = e >> 6;
  vT[(((size_t)(b * HEADS + h) * DK) + d) * SEQ + s] =
      y[(size_t)(b * SEQ + s) * DMODEL + e];
}

// ---------------------------------------------------------------------------
// Causal flash attention. One wave per (b,h,qtile of 16 rows).
// Per 32-key chunk: 4 WMMAs for scores (dk=64), online softmax (f32) with
//   - row-max via half-wave butterflies (ds_bpermute),
//   - row-sum fused into the matrix pipe: l-accumulator tile = P · ones,
// P re-layout through LDS (DS ops are in-order within a wave; single-wave
// workgroup so no barrier), then 4 WMMAs for P·V.
// __launch_bounds__(32,1): occupancy is irrelevant for a 1-wave WG; give the
// scheduler the full VGPR budget so Q frags / accumulators never spill.
// ---------------------------------------------------------------------------
__global__ void __launch_bounds__(32, 1) flash_attn(
    const __bf16* __restrict__ q, const __bf16* __restrict__ k,
    const __bf16* __restrict__ vT, __bf16* __restrict__ o) {
  __shared__ __bf16 Pbuf[16 * 32];

  const int lane  = threadIdx.x & 31;
  const int gw    = blockIdx.x;        // (b*H+h)*128 + qtile
  const int qtile = gw & 127;          // S/16 = 128
  const int bh    = gw >> 7;
  const int qbase = qtile * 16;
  const int n = lane & 15, half = lane >> 4;

  const __bf16* qp = q  + ((size_t)bh * SEQ + qbase) * DK;
  const __bf16* kp = k  + (size_t)bh * SEQ * DK;
  const __bf16* vp = vT + (size_t)bh * DK * SEQ;

  const bf16x16 qa0 = load_a_frag(qp +  0, DK);
  const bf16x16 qa1 = load_a_frag(qp + 32, DK);

  bf16x16 ones;
#pragma unroll
  for (int j = 0; j < 16; ++j) ones[j] = __builtin_bit_cast(__bf16, (unsigned short)0x3F80);

  v8f o0 = {}, o1 = {}, o2 = {}, o3 = {}, lacc = {};
  float mrow[8];
#pragma unroll
  for (int r = 0; r < 8; ++r) mrow[r] = -INFINITY;

  for (int kc = 0; kc <= qbase + 15; kc += 32) {
    // ---- scores: S[16 x 32] = Q(16x64) · K(32x64)ᵀ, 4 WMMAs ----
    bf16x16 kb00 = load_bT_frag(kp + (size_t)kc * DK +  0, DK);
    bf16x16 kb01 = load_bT_frag(kp + (size_t)kc * DK + 32, DK);
    bf16x16 kb10 = load_bT_frag(kp + (size_t)(kc + 16) * DK +  0, DK);
    bf16x16 kb11 = load_bT_frag(kp + (size_t)(kc + 16) * DK + 32, DK);
    v8f s0 = {}, s1 = {};
    s0 = wmma_bf16(qa0, kb00, s0);
    s0 = wmma_bf16(qa1, kb01, s0);
    s1 = wmma_bf16(qa0, kb10, s1);
    s1 = wmma_bf16(qa1, kb11, s1);

    // ---- online softmax: mask, row-max butterfly, exp -> LDS (bf16) ----
    const int key0 = kc + n, key1 = kc + 16 + n;
#pragma unroll
    for (int r = 0; r < 8; ++r) {
      const int qrow = qbase + r + 8 * half;
      const float a = (key0 <= qrow) ? s0[r] * SCALE : -INFINITY;
      const float b = (key1 <= qrow) ? s1[r] * SCALE : -INFINITY;
      float rm = fmaxf(a, b);
#pragma unroll
      for (int off = 1; off < 16; off <<= 1)
        rm = fmaxf(rm, __shfl_xor(rm, off, 32));
      const float mn   = fmaxf(mrow[r], rm);
      const float corr = __expf(mrow[r] - mn);
      mrow[r] = mn;
      const int row = r + 8 * half;
      Pbuf[row * 32 +      n] = to_bf16(__expf(a - mn));
      Pbuf[row * 32 + 16 + n] = to_bf16(__expf(b - mn));
      o0[r] *= corr; o1[r] *= corr; o2[r] *= corr; o3[r] *= corr;
      lacc[r] *= corr;
    }
    const bf16x16 pa = load_a_frag(Pbuf, 32);  // DS ops in-order within wave

    // ---- row-sum through the matrix pipe: l-tile += P · ones ----
    lacc = wmma_bf16(pa, ones, lacc);

    // ---- O += P(16x32) · V(32x64): vT rows are contiguous in keys ----
    o0 = wmma_bf16(pa, load_bT_frag(vp + (size_t) 0 * SEQ + kc, SEQ), o0);
    o1 = wmma_bf16(pa, load_bT_frag(vp + (size_t)16 * SEQ + kc, SEQ), o1);
    o2 = wmma_bf16(pa, load_bT_frag(vp + (size_t)32 * SEQ + kc, SEQ), o2);
    o3 = wmma_bf16(pa, load_bT_frag(vp + (size_t)48 * SEQ + kc, SEQ), o3);
  }

  // ---- epilogue: normalize, write o[b, s, h*dk + d] ----
  const int b = bh >> 4, h = bh & 15;
#pragma unroll
  for (int r = 0; r < 8; ++r) {
    const int qrow = qbase + r + 8 * half;
    const float inv = 1.0f / lacc[r];   // every column of l-tile = rowsum
    __bf16* op = o + ((size_t)(b * SEQ + qrow)) * DMODEL + h * DK;
    op[     n] = to_bf16(o0[r] * inv);
    op[16 + n] = to_bf16(o1[r] * inv);
    op[32 + n] = to_bf16(o2[r] * inv);
    op[48 + n] = to_bf16(o3[r] * inv);
  }
}

// ---------------------------------------------------------------------------
// Host-side orchestration (all on `stream`, graph-capture safe).
// ---------------------------------------------------------------------------
extern "C" void kernel_launch(void* const* d_in, const int* in_sizes, int n_in,
                              void* d_out, int out_size, void* d_ws, size_t ws_size,
                              hipStream_t stream) {
  (void)in_sizes; (void)n_in; (void)out_size; (void)ws_size;

  const float* x    = (const float*)d_in[0];
  const float* Wq   = (const float*)d_in[1];
  const float* Wk   = (const float*)d_in[2];
  const float* Wv   = (const float*)d_in[3];
  const float* Wo   = (const float*)d_in[4];
  const float* cosT = (const float*)d_in[5];
  const float* sinT = (const float*)d_in[6];
  float* out = (float*)d_out;

  const int BS   = BATCH * SEQ;          // 4096
  const int NX   = BS * DMODEL;          // 4,194,304
  const int NW   = DMODEL * DMODEL;      // 1,048,576

  // Workspace layout (bf16 buffers), ~56 MB total
  char* base = (char*)d_ws;
  __bf16* xb   = (__bf16*)(base + (size_t) 0 * 1024 * 1024);  //  8 MB
  __bf16* wqb  = (__bf16*)(base + (size_t) 8 * 1024 * 1024);  //  2 MB
  __bf16* wkb  = (__bf16*)(base + (size_t)10 * 1024 * 1024);  //  2 MB
  __bf16* wvb  = (__bf16*)(base + (size_t)12 * 1024 * 1024);  //  2 MB
  __bf16* wob  = (__bf16*)(base + (size_t)14 * 1024 * 1024);  //  2 MB
  __bf16* ybuf = (__bf16*)(base + (size_t)16 * 1024 * 1024);  //  8 MB (reused)
  __bf16* qb   = (__bf16*)(base + (size_t)24 * 1024 * 1024);  //  8 MB
  __bf16* kb   = (__bf16*)(base + (size_t)32 * 1024 * 1024);  //  8 MB
  __bf16* vTb  = (__bf16*)(base + (size_t)40 * 1024 * 1024);  //  8 MB
  __bf16* ob   = (__bf16*)(base + (size_t)48 * 1024 * 1024);  //  8 MB

  // 1) fp32 -> bf16
  cvt_f32_bf16<<<2048, 256, 0, stream>>>(x,  xb,  NX);
  cvt_f32_bf16<<<1024, 256, 0, stream>>>(Wq, wqb, NW);
  cvt_f32_bf16<<<1024, 256, 0, stream>>>(Wk, wkb, NW);
  cvt_f32_bf16<<<1024, 256, 0, stream>>>(Wv, wvb, NW);
  cvt_f32_bf16<<<1024, 256, 0, stream>>>(Wo, wob, NW);

  // GEMM grid: (M/32)*(N/64) waves, 4 waves (128 thr) per block
  const int gemm_blocks = ((BS / 32) * (DMODEL / 64)) / 4;   // 512

  // 2) Q projection + RoPE
  gemm_xwt<__bf16><<<gemm_blocks, 128, 0, stream>>>(xb, wqb, ybuf, BS, DMODEL, DMODEL);
  rope_permute<<<(BATCH * SEQ * HEADS * 32) / 256, 256, 0, stream>>>(ybuf, cosT, sinT, qb);

  // 3) K projection + RoPE
  gemm_xwt<__bf16><<<gemm_blocks, 128, 0, stream>>>(xb, wkb, ybuf, BS, DMODEL, DMODEL);
  rope_permute<<<(BATCH * SEQ * HEADS * 32) / 256, 256, 0, stream>>>(ybuf, cosT, sinT, kb);

  // 4) V projection + transpose to [B,H,dk,S]
  gemm_xwt<__bf16><<<gemm_blocks, 128, 0, stream>>>(xb, wvb, ybuf, BS, DMODEL, DMODEL);
  permute_vT<<<NX / 256, 256, 0, stream>>>(ybuf, vTb);

  // 5) causal flash attention: one wave per 16-row Q tile
  flash_attn<<<BATCH * HEADS * (SEQ / 16), 32, 0, stream>>>(qb, kb, vTb, ob);

  // 6) output projection, f32 result
  gemm_xwt<float><<<gemm_blocks, 128, 0, stream>>>(ob, wob, out, BS, DMODEL, DMODEL);
}
